// TrivialTreeTagger_5669356831080
// MI455X (gfx1250) — matrix operand order repaired and checked by reading, not compiled, not run on hardware
//
#include <hip/hip_runtime.h>
#include <hip/hip_bf16.h>

// MI455X (gfx1250) tree-LSTM tagger.
// Roofline: ~33 GFLOP vs ~75MB traffic -> compute bound -> bf16 WMMA
// (v_wmma_f32_16x16x32_bf16), f32 accumulate. One wave32 = 16 batch rows.

typedef __attribute__((ext_vector_type(16))) __bf16 v16bf;
typedef __attribute__((ext_vector_type(8)))  __bf16 v8bf;
typedef __attribute__((ext_vector_type(8)))  float  v8f;

#define WAVES 8
#define RS_ST 80            // bf16 staging row stride (elements), 160B rows (16B aligned)
#define LB_ST 72            // f32 logits row stride
#define SCR_F (16*LB_ST+16) // floats of per-wave scratch (staging/logits/rowstats overlap in time)

union U16 { v16bf v; v8bf h[2]; };

__device__ __forceinline__ v8f wmma_bf16(v16bf a, v16bf b, v8f c) {
  return __builtin_amdgcn_wmma_f32_16x16x32_bf16(false, a, false, b, (short)0, c, false, false);
}

__device__ __forceinline__ v8f splat8(float x) {
  v8f v;
  #pragma unroll
  for (int i = 0; i < 8; i++) v[i] = x;
  return v;
}

// Branch-free transcendentals: v_exp_f32 + v_rcp_f32, no EXEC games.
__device__ __forceinline__ float sigm(float x) {
  return __builtin_amdgcn_rcpf(1.0f + __expf(-x));
}
__device__ __forceinline__ float tanh_fast(float x) {
  // tanh(x) = 1 - 2/(1+e^{2x}); exact limits at +-inf, no NaN path.
  return 1.0f - 2.0f * __builtin_amdgcn_rcpf(1.0f + __expf(2.0f * x));
}

// B fragment (32x16 bf16) from LDS stored as [N][K]: lane holds N=lane%16,
// K-chunks at c0 and c0+16 (c0 = k0 + (lane<16?0:8)) -> two 16B ds loads.
__device__ __forceinline__ v16bf load_bfrag(const __bf16* W, int rs, int n0, int k0, int lane) {
  int n = n0 + (lane & 15);
  int k = k0 + ((lane & 16) ? 8 : 0);
  const __bf16* p = W + n * rs + k;
  U16 u;
  u.h[0] = *(const v8bf*)(p);
  u.h[1] = *(const v8bf*)(p + 16);
  return u.v;
}

// A fragment (16x32 bf16) from row-major bf16 staging buffer [16][RS_ST]:
// lane holds row=lane%16, K-chunks at c0 and c0+16 (c0 = kt*32 + (lane<16?0:8)).
__device__ __forceinline__ v16bf load_afrag_lds(const __bf16* st, int ktile, int lane) {
  int row = lane & 15;
  int c0 = ktile * 32 + ((lane & 16) ? 8 : 0);
  const __bf16* p = st + row * RS_ST + c0;
  U16 u;
  u.h[0] = *(const v8bf*)(p);
  u.h[1] = *(const v8bf*)(p + 16);
  return u.v;
}

// A fragment straight from the f32 embedding table (gather by token id).
__device__ __forceinline__ v16bf load_afrag_emb(const float* __restrict__ emb,
                                                const int* __restrict__ x,
                                                int m0, int t, int lane) {
  int row = lane & 15;
  int tok = x[(m0 + row) * 16 + t];
  int c0 = (lane & 16) ? 8 : 0;
  const float* e = emb + (long)tok * 32 + c0;
  v16bf v;
  #pragma unroll
  for (int i = 0; i < 8; i++) v[i] = (__bf16)e[i];
  #pragma unroll
  for (int i = 0; i < 8; i++) v[8 + i] = (__bf16)e[16 + i];
  return v;
}

__device__ __forceinline__ void stage_tile(__bf16* st, int ntile, v8f c, int lane) {
  int col = ntile * 16 + (lane & 15);
  int r0 = (lane & 16) ? 8 : 0;
  #pragma unroll
  for (int r = 0; r < 8; r++) st[(r0 + r) * RS_ST + col] = (__bf16)c[r];
}

__global__ __launch_bounds__(256) void tree_tagger_kernel(
    const int* __restrict__ x, const int* __restrict__ s,
    const float* __restrict__ emb,
    const float* __restrict__ Wih, const float* __restrict__ Whh,
    const float* __restrict__ bih, const float* __restrict__ bhh,
    const float* __restrict__ h0, const float* __restrict__ c0,
    const float* __restrict__ Wc, const float* __restrict__ bc,
    const float* __restrict__ Wt, const float* __restrict__ bt,
    float* __restrict__ out)
{
  // Weights in LDS, bf16, laid out [N][K] for direct B-fragment loads.
  __shared__ __align__(16) __bf16 sWg[256][96]; // [Wih | Whh] gates, K=96
  __shared__ __align__(16) __bf16 sWcm[32][96]; // combiner, K=96
  __shared__ __align__(16) __bf16 sWtg[64][32]; // tag proj, K=32
  __shared__ float sBg[256];
  __shared__ float sBc[32];
  __shared__ float sBt[64];
  __shared__ __align__(16) float sScr[WAVES][SCR_F]; // per-wave transpose/logits scratch

  const int tid = threadIdx.x;

  { // cooperative weight preload, f32 -> bf16
    int n = tid; // 0..255
    #pragma unroll 4
    for (int k = 0; k < 32; k++) sWg[n][k] = (__bf16)Wih[n * 32 + k];
    #pragma unroll 4
    for (int k = 0; k < 64; k++) sWg[n][32 + k] = (__bf16)Whh[n * 64 + k];
    sBg[n] = bih[n] + bhh[n];
    if (n < 32) {
      #pragma unroll 4
      for (int k = 0; k < 96; k++) sWcm[n][k] = (__bf16)Wc[n * 96 + k];
      sBc[n] = bc[n];
    }
    if (n < 64) {
      #pragma unroll 4
      for (int k = 0; k < 32; k++) sWtg[n][k] = (__bf16)Wt[n * 32 + k];
      sBt[n] = bt[n];
    }
  }

  const int lane  = tid & 31;
  const int wave  = tid >> 5;
  const int nlane = lane & 15;
  const int r0    = (lane & 16) ? 8 : 0;
  const int m0    = (blockIdx.x * WAVES + wave) * 16;

  __bf16* st    = (__bf16*)&sScr[wave][0];
  float*  lb    = &sScr[wave][0];
  float*  rstat = &sScr[wave][16 * LB_ST];

  __builtin_prefetch(&x[m0 * 16], 0, 0); // global_prefetch_b8 for token ids

  // mask for last timestep: only o[:, -1] survives -> (s >= 4)
  float mk[8];
  #pragma unroll
  for (int r = 0; r < 8; r++) mk[r] = (s[m0 + r0 + r] >= 4) ? 1.0f : 0.0f;

  // initial-h A-fragments (every row == h0) and per-lane c0
  v16bf hinit0, hinit1;
  {
    int koff = (lane & 16) ? 8 : 0;
    #pragma unroll
    for (int i = 0; i < 16; i++) {
      int kk = koff + ((i < 8) ? i : (i + 8));
      hinit0[i] = (__bf16)h0[kk];
      hinit1[i] = (__bf16)h0[32 + kk];
    }
  }
  float c0v[4];
  #pragma unroll
  for (int hc = 0; hc < 4; hc++) c0v[hc] = c0[hc * 16 + nlane];

  __syncthreads();

  // per-lane bias registers (accumulators start from inline-zero C instead)
  float bg[16]; // [q*4+hc]
  #pragma unroll
  for (int q = 0; q < 4; q++)
    #pragma unroll
    for (int hc = 0; hc < 4; hc++) bg[q * 4 + hc] = sBg[(q * 4 + hc) * 16 + nlane];
  float bcv0 = sBc[nlane], bcv1 = sBc[16 + nlane];
  float btv[4];
  #pragma unroll
  for (int nt = 0; nt < 4; nt++) btv[nt] = sBt[nt * 16 + nlane];

  const v8f zc = splat8(0.0f); // -> inline SRC2=0 in first WMMA of each chain

  v16bf cF0, cF1, cF2, cF3, cF4; // blueprint hidden states (combs), A-layout bf16

  #pragma clang loop unroll(disable)
  for (int b = 0; b < 5; b++) {
    v16bf aC0, aC1, aC2, aC3, aR;
    if (b < 4) {
      aC0 = load_afrag_emb(emb, x, m0, 4 * b + 0, lane);
      aC1 = load_afrag_emb(emb, x, m0, 4 * b + 1, lane);
      aC2 = load_afrag_emb(emb, x, m0, 4 * b + 2, lane);
      aC3 = load_afrag_emb(emb, x, m0, 4 * b + 3, lane);
      aR  = aC0; // root token == first child token for these blueprints
    } else {
      aC0 = cF0; aC1 = cF1; aC2 = cF2; aC3 = cF3;
      aR  = load_afrag_emb(emb, x, m0, 0, lane);
    }

    v16bf aH0 = hinit0, aH1 = hinit1;
    v8f cst[4];
    #pragma unroll
    for (int hc = 0; hc < 4; hc++) cst[hc] = splat8(c0v[hc]);

    #pragma clang loop unroll(disable)
    for (int t = 0; t < 4; t++) {
      v16bf aX = (t == 0) ? aC0 : ((t == 1) ? aC1 : ((t == 2) ? aC2 : aC3));
      __syncthreads(); // staging buffer free for rewrite
      #pragma unroll
      for (int hc = 0; hc < 4; hc++) {
        v8f ai = zc, af = zc, ag = zc, ao = zc;
        #pragma unroll
        for (int kt = 0; kt < 3; kt++) {
          v16bf a = (kt == 0) ? aX : ((kt == 1) ? aH0 : aH1);
          ai = wmma_bf16(a, load_bfrag(&sWg[0][0], 96, (0 * 4 + hc) * 16, kt * 32, lane), ai);
          af = wmma_bf16(a, load_bfrag(&sWg[0][0], 96, (1 * 4 + hc) * 16, kt * 32, lane), af);
          ag = wmma_bf16(a, load_bfrag(&sWg[0][0], 96, (2 * 4 + hc) * 16, kt * 32, lane), ag);
          ao = wmma_bf16(a, load_bfrag(&sWg[0][0], 96, (3 * 4 + hc) * 16, kt * 32, lane), ao);
        }
        int col = hc * 16 + nlane;
        #pragma unroll
        for (int r = 0; r < 8; r++) {
          float iv = sigm(ai[r] + bg[0 * 4 + hc]);
          float fv = sigm(af[r] + bg[1 * 4 + hc]);
          float gv = tanh_fast(ag[r] + bg[2 * 4 + hc]);
          float ov = sigm(ao[r] + bg[3 * 4 + hc]);
          float cn = fv * cst[hc][r] + iv * gv;
          cst[hc][r] = cn;
          float hn = ov * tanh_fast(cn);
          float val = (t == 3) ? fmaxf(hn, 0.0f) * mk[r] : hn;
          st[(r0 + r) * RS_ST + col] = (__bf16)val; // C-layout -> staging
        }
      }
      __syncthreads();
      aH0 = load_afrag_lds(st, 0, lane); // staging -> A-layout
      aH1 = load_afrag_lds(st, 1, lane);
    }
    // after t-loop: aH0/aH1 hold "last" = relu(h3)*mask fragments

    // comb = relu([root, last] @ Wc^T + bc)   (16x96 @ 96x32)
    v8f cc0 = zc, cc1 = zc;
    #pragma unroll
    for (int kt = 0; kt < 3; kt++) {
      v16bf a = (kt == 0) ? aR : ((kt == 1) ? aH0 : aH1);
      cc0 = wmma_bf16(a, load_bfrag(&sWcm[0][0], 96, 0, kt * 32, lane), cc0);
      cc1 = wmma_bf16(a, load_bfrag(&sWcm[0][0], 96, 16, kt * 32, lane), cc1);
    }
    #pragma unroll
    for (int i = 0; i < 8; i++) {
      cc0[i] = fmaxf(cc0[i] + bcv0, 0.0f);
      cc1[i] = fmaxf(cc1[i] + bcv1, 0.0f);
    }
    __syncthreads();
    stage_tile(st, 0, cc0, lane);
    stage_tile(st, 1, cc1, lane);
    __syncthreads();
    v16bf cf = load_afrag_lds(st, 0, lane);
    if      (b == 0) cF0 = cf;
    else if (b == 1) cF1 = cf;
    else if (b == 2) cF2 = cf;
    else if (b == 3) cF3 = cf;
    else             cF4 = cf;
  }

  // logits = comb4 @ Wt^T + bt   (16x32 @ 32x64)
  v8f lacc[4];
  #pragma unroll
  for (int nt = 0; nt < 4; nt++)
    lacc[nt] = wmma_bf16(cF4, load_bfrag(&sWtg[0][0], 32, nt * 16, 0, lane), zc);

  // log_softmax over 64 tags per row
  __syncthreads();
  #pragma unroll
  for (int nt = 0; nt < 4; nt++) {
    int col = nt * 16 + nlane;
    #pragma unroll
    for (int r = 0; r < 8; r++) lb[(r0 + r) * LB_ST + col] = lacc[nt][r] + btv[nt];
  }
  __syncthreads();
  if (lane < 16) {
    const float* row = &lb[lane * LB_ST];
    float mx = row[0];
    for (int c = 1; c < 64; c++) mx = fmaxf(mx, row[c]);
    float sum = 0.0f;
    for (int c = 0; c < 64; c++) sum += __expf(row[c] - mx);
    rstat[lane] = mx + __logf(sum);
  }
  __syncthreads();
  #pragma unroll 4
  for (int i = 0; i < 32; i++) {       // coalesced output: 16 rows x 64 cols
    int idx = i * 32 + lane;
    int rr = idx >> 6;
    int cc = idx & 63;
    out[(long)(m0 + rr) * 64 + cc] = lb[rr * LB_ST + cc] - rstat[rr];
  }
}

extern "C" void kernel_launch(void* const* d_in, const int* in_sizes, int n_in,
                              void* d_out, int out_size, void* d_ws, size_t ws_size,
                              hipStream_t stream) {
  const int*   x   = (const int*)d_in[0];
  const int*   s   = (const int*)d_in[1];
  const float* emb = (const float*)d_in[2];
  const float* Wih = (const float*)d_in[3];
  const float* Whh = (const float*)d_in[4];
  const float* bih = (const float*)d_in[5];
  const float* bhh = (const float*)d_in[6];
  const float* h0  = (const float*)d_in[7];
  const float* c0  = (const float*)d_in[8];
  const float* Wc  = (const float*)d_in[9];
  const float* bc  = (const float*)d_in[10];
  const float* Wt  = (const float*)d_in[11];
  const float* bt  = (const float*)d_in[12];
  float* out = (float*)d_out;

  int B = in_sizes[0] / 16;              // 32768
  int blocks = B / (16 * WAVES);         // 16 rows per wave, 8 waves per block
  hipLaunchKernelGGL(tree_tagger_kernel, dim3(blocks), dim3(256), 0, stream,
                     x, s, emb, Wih, Whh, bih, bhh, h0, c0, Wc, bc, Wt, bt, out);
}
